// Processor_24172075942170
// MI455X (gfx1250) — matrix operand order repaired and checked by reading, not compiled, not run on hardware
//
#include <hip/hip_runtime.h>

#define N_NODES 50000
#define N_EDGES 300000
#define DIM 256

typedef __attribute__((ext_vector_type(16))) __bf16 v16bf;
typedef __attribute__((ext_vector_type(8)))  __bf16 v8bf;
typedef __attribute__((ext_vector_type(8)))  float  v8f;

extern __shared__ char smem[];

__device__ __forceinline__ v8f wmma_bf16(v16bf a, v16bf b, v8f c) {
    return __builtin_amdgcn_wmma_f32_16x16x32_bf16(
        /*neg_a=*/false, a, /*neg_b=*/false, b,
        /*c_mod=*/(short)0, c, /*reuse_a=*/false, /*reuse_b=*/false);
}

// A fragment: 16x32 bf16 tile from LDS, row-major, leading dim ldk.
// Lanes 0-15: M=lane, K = k0 + {0..7, 16..23}; lanes 16-31: K = k0 + {8..15, 24..31}.
__device__ __forceinline__ v16bf load_a_frag(const __bf16* A, int ldk, int k0, int lane) {
    int m = lane & 15, kb = (lane >> 4) * 8;
    const __bf16* p = A + m * ldk + k0 + kb;
    v8bf lo = *(const v8bf*)p;
    v8bf hi = *(const v8bf*)(p + 16);
    v16bf a;
#pragma unroll
    for (int j = 0; j < 8; ++j) { a[j] = lo[j]; a[j + 8] = hi[j]; }
    return a;
}

// B fragment: weights transposed [N][K]; lane n = n0+(lane&15) reads
// K = k0 + {0..7, 16..23} (lanes 0-15) or k0 + {8..15, 24..31} (lanes 16-31).
__device__ __forceinline__ v16bf load_b_frag(const __bf16* Wt, int ldk, int n0, int k0, int lane) {
    int n = lane & 15, kb = (lane >> 4) * 8;
    const __bf16* p = Wt + (size_t)(n0 + n) * ldk + k0 + kb;
    v8bf lo = *(const v8bf*)p;
    v8bf hi = *(const v8bf*)(p + 16);
    v16bf b;
#pragma unroll
    for (int j = 0; j < 8; ++j) { b[j] = lo[j]; b[j + 8] = hi[j]; }
    return b;
}

// One MLP layer over a 64-row tile: [64,K] (LDS bf16) x [K,256] -> [64,256].
// 8 waves; each wave owns two 16-wide N-tiles across 4 M-tiles (8 accumulators),
// so every B fragment fetched from L2 is reused 4x in registers.
template <int K>
__device__ __forceinline__ void gemm_layer64(const __bf16* Ain, const __bf16* Wt,
                                             const float* bias, bool relu,
                                             __bf16* Hout, float* Fout) {
    int tid = threadIdx.x;
    int wave = tid >> 5, lane = tid & 31;
    int n0 = wave * 32;
    v8f acc[4][2] = {};
    for (int k0 = 0; k0 < K; k0 += 32) {
        v16bf b0 = load_b_frag(Wt, K, n0,      k0, lane);
        v16bf b1 = load_b_frag(Wt, K, n0 + 16, k0, lane);
#pragma unroll
        for (int mt = 0; mt < 4; ++mt) {
            v16bf a = load_a_frag(Ain + mt * 16 * K, K, k0, lane);
            acc[mt][0] = wmma_bf16(a, b0, acc[mt][0]);
            acc[mt][1] = wmma_bf16(a, b1, acc[mt][1]);
        }
    }
    int half = lane >> 4;
    int nA = n0 + (lane & 15), nB = nA + 16;
    float bA = bias[nA], bB = bias[nB];
#pragma unroll
    for (int mt = 0; mt < 4; ++mt) {
#pragma unroll
        for (int j = 0; j < 8; ++j) {
            int m = mt * 16 + j + half * 8;  // C layout: VGPR j -> M=j / j+8
            float vA = acc[mt][0][j] + bA, vB = acc[mt][1][j] + bB;
            if (relu) { vA = fmaxf(vA, 0.f); vB = fmaxf(vB, 0.f); }
            if (Hout) { Hout[m * 256 + nA] = (__bf16)vA; Hout[m * 256 + nB] = (__bf16)vB; }
            if (Fout) { Fout[m * 256 + nA] = vA; Fout[m * 256 + nB] = vB; }
        }
    }
}

// Shared LayerNorm stats over F[64][256] using 256 threads.
__device__ __forceinline__ void ln_stats(const float* F, float* Ps, float* Pq,
                                         float* Mu, float* Rs) {
    int tid = threadIdx.x;
    int r = tid >> 2, cb = (tid & 3) * 64;
    float s = 0.f, ss = 0.f;
    for (int c = cb; c < cb + 64; ++c) { float v = F[r * 256 + c]; s += v; ss += v * v; }
    Ps[tid] = s; Pq[tid] = ss;
    __syncthreads();
    if (tid < 64) {
        float s2 = 0.f, ss2 = 0.f;
#pragma unroll
        for (int j = 0; j < 4; ++j) { s2 += Ps[tid * 4 + j]; ss2 += Pq[tid * 4 + j]; }
        float mu = s2 * (1.f / 256.f);
        float var = ss2 * (1.f / 256.f) - mu * mu;
        Mu[tid] = mu; Rs[tid] = rsqrtf(var + 1e-5f);
    }
    __syncthreads();
}

// LDS layout (dynamic, 133632 B):
//   edge: Ain 64x768 bf16 [0,98304) | H1 64x256 bf16 [98304,131072)
//         H2 aliases [0,32768), F(f32) aliases [32768,98304) | redn [131072,133632)
//   node: Ain 64x512 bf16 [0,65536) | H1 [65536,98304) | H2 [98304,131072)
//         F aliases Ain [0,65536) | redn [131072,133632)
#define SMEM_BYTES 133632

// ---------- Edge update: ea += LN(MLP(concat(x[row], x[col], ea))); agg[col] += ea_new ----------
__global__ __launch_bounds__(256) void edge_kernel(
    const float* __restrict__ x, float* __restrict__ ea,
    const long long* __restrict__ rowI, const long long* __restrict__ colI,
    const __bf16* __restrict__ W0t, const float* __restrict__ b0,
    const __bf16* __restrict__ W1t, const float* __restrict__ b1,
    const __bf16* __restrict__ W2t, const float* __restrict__ b2,
    const float* __restrict__ gam, const float* __restrict__ bet,
    float* __restrict__ agg)
{
    __bf16* Ain = (__bf16*)smem;
    __bf16* H1  = (__bf16*)(smem + 98304);
    __bf16* H2  = (__bf16*)smem;             // alias (Ain dead after layer0)
    float*  F   = (float*)(smem + 32768);    // alias
    float*  Ps  = (float*)(smem + 131072);
    float*  Pq  = Ps + 256;
    float*  Mu  = Pq + 256;
    float*  Rs  = Mu + 64;

    int tid = threadIdx.x;
    int e0 = blockIdx.x * 64;

    // Gather + f32->bf16 convert: 64 x 768 tile, float4 granularity
    for (int idx = tid; idx < 64 * 192; idx += 256) {
        int r = idx / 192, c = (idx - r * 192) * 4;
        int e = e0 + r; if (e >= N_EDGES) e = N_EDGES - 1;  // clamped read, writes guarded later
        const float* src;
        if (c < 256)       src = x  + (size_t)rowI[e] * 256 + c;
        else if (c < 512)  src = x  + (size_t)colI[e] * 256 + (c - 256);
        else               src = ea + (size_t)e       * 256 + (c - 512);
        float4 v = *(const float4*)src;
        __bf16* d = Ain + (size_t)r * 768 + c;
        d[0] = (__bf16)v.x; d[1] = (__bf16)v.y; d[2] = (__bf16)v.z; d[3] = (__bf16)v.w;
    }
    __syncthreads();

    gemm_layer64<768>(Ain, W0t, b0, true, H1, nullptr);
    __syncthreads();
    gemm_layer64<256>(H1, W1t, b1, true, H2, nullptr);
    __syncthreads();
    gemm_layer64<256>(H2, W2t, b2, false, nullptr, F);
    __syncthreads();

    ln_stats(F, Ps, Pq, Mu, Rs);

    int r = tid >> 2, cb = (tid & 3) * 64;
    int e = e0 + r;
    if (e < N_EDGES) {
        float mu = Mu[r], rs = Rs[r];
        long long cd = colI[e];
        float* earow  = ea  + (size_t)e  * 256;
        float* aggrow = agg + (size_t)cd * 256;
        for (int c = cb; c < cb + 64; ++c) {
            float o  = (F[r * 256 + c] - mu) * rs * gam[c] + bet[c];
            float nv = earow[c] + o;      // residual
            earow[c] = nv;
            atomicAdd(&aggrow[c], nv);    // fused segment_sum
        }
    }
}

// ---------- Node update: x += LN(MLP(concat(x, agg))) ----------
__global__ __launch_bounds__(256) void node_kernel(
    float* __restrict__ x, const float* __restrict__ agg,
    const __bf16* __restrict__ W0t, const float* __restrict__ b0,
    const __bf16* __restrict__ W1t, const float* __restrict__ b1,
    const __bf16* __restrict__ W2t, const float* __restrict__ b2,
    const float* __restrict__ gam, const float* __restrict__ bet)
{
    __bf16* Ain = (__bf16*)smem;
    __bf16* H1  = (__bf16*)(smem + 65536);
    __bf16* H2  = (__bf16*)(smem + 98304);
    float*  F   = (float*)smem;              // alias Ain (dead after layer0)
    float*  Ps  = (float*)(smem + 131072);
    float*  Pq  = Ps + 256;
    float*  Mu  = Pq + 256;
    float*  Rs  = Mu + 64;

    int tid = threadIdx.x;
    int v0 = blockIdx.x * 64;

    for (int idx = tid; idx < 64 * 128; idx += 256) {
        int r = idx >> 7, c = (idx & 127) * 4;
        int v = v0 + r; if (v >= N_NODES) v = N_NODES - 1;
        const float* src = (c < 256) ? x + (size_t)v * 256 + c
                                     : agg + (size_t)v * 256 + (c - 256);
        float4 val = *(const float4*)src;
        __bf16* d = Ain + (size_t)r * 512 + c;
        d[0] = (__bf16)val.x; d[1] = (__bf16)val.y; d[2] = (__bf16)val.z; d[3] = (__bf16)val.w;
    }
    __syncthreads();

    gemm_layer64<512>(Ain, W0t, b0, true, H1, nullptr);
    __syncthreads();
    gemm_layer64<256>(H1, W1t, b1, true, H2, nullptr);
    __syncthreads();
    gemm_layer64<256>(H2, W2t, b2, false, nullptr, F);
    __syncthreads();

    ln_stats(F, Ps, Pq, Mu, Rs);

    int r = tid >> 2, cb = (tid & 3) * 64;
    int v = v0 + r;
    if (v < N_NODES) {
        float mu = Mu[r], rs = Rs[r];
        float* xrow = x + (size_t)v * 256;
        for (int c = cb; c < cb + 64; ++c) {
            float o = (F[r * 256 + c] - mu) * rs * gam[c] + bet[c];
            xrow[c] += o;
        }
    }
}

// Transpose + convert weights: src [9][K][256] f32 -> dst [9][256][K] bf16
__global__ __launch_bounds__(256) void wcvt_kernel(const float* __restrict__ src,
                                                   __bf16* __restrict__ dst,
                                                   int K, long long total) {
    long long i = (long long)blockIdx.x * 256 + threadIdx.x;
    if (i >= total) return;
    long long KN = (long long)K * 256;
    int b = (int)(i / KN);
    long long rem = i - (long long)b * KN;
    int n = (int)(rem / K);
    int k = (int)(rem - (long long)n * K);
    dst[i] = (__bf16)src[(long long)b * KN + (long long)k * 256 + n];
}

__global__ __launch_bounds__(256) void zero_kernel(float* __restrict__ p, long long n) {
    long long i = (long long)blockIdx.x * 256 + threadIdx.x;
    if (i < n) p[i] = 0.f;
}

extern "C" void kernel_launch(void* const* d_in, const int* in_sizes, int n_in,
                              void* d_out, int out_size, void* d_ws, size_t ws_size,
                              hipStream_t stream) {
    const float*     x_in  = (const float*)d_in[0];
    const long long* eidx  = (const long long*)d_in[1];
    const float*     ea_in = (const float*)d_in[2];
    const float* eW0 = (const float*)d_in[3];  const float* eb0 = (const float*)d_in[4];
    const float* eW1 = (const float*)d_in[5];  const float* eb1 = (const float*)d_in[6];
    const float* eW2 = (const float*)d_in[7];  const float* eb2 = (const float*)d_in[8];
    const float* eg  = (const float*)d_in[9];  const float* ebt = (const float*)d_in[10];
    const float* nW0 = (const float*)d_in[11]; const float* nb0 = (const float*)d_in[12];
    const float* nW1 = (const float*)d_in[13]; const float* nb1 = (const float*)d_in[14];
    const float* nW2 = (const float*)d_in[15]; const float* nb2 = (const float*)d_in[16];
    const float* ng  = (const float*)d_in[17]; const float* nbt = (const float*)d_in[18];

    char* ws = (char*)d_ws;
    size_t off = 0;
    float* ea  = (float*)(ws + off); off += (size_t)N_EDGES * DIM * 4;
    float* agg = (float*)(ws + off); off += (size_t)N_NODES * DIM * 4;
    __bf16* eW0t = (__bf16*)(ws + off); off += 9ull * 768 * 256 * 2;
    __bf16* eW1t = (__bf16*)(ws + off); off += 9ull * 256 * 256 * 2;
    __bf16* eW2t = (__bf16*)(ws + off); off += 9ull * 256 * 256 * 2;
    __bf16* nW0t = (__bf16*)(ws + off); off += 9ull * 512 * 256 * 2;
    __bf16* nW1t = (__bf16*)(ws + off); off += 9ull * 256 * 256 * 2;
    __bf16* nW2t = (__bf16*)(ws + off); off += 9ull * 256 * 256 * 2;

    float* xbuf = (float*)d_out;  // working x; final output lands here

    hipMemcpyAsync(xbuf, x_in,  (size_t)N_NODES * DIM * 4, hipMemcpyDeviceToDevice, stream);
    hipMemcpyAsync(ea,   ea_in, (size_t)N_EDGES * DIM * 4, hipMemcpyDeviceToDevice, stream);

    auto cvt = [&](const float* s, __bf16* d, int K) {
        long long tot = 9ll * K * 256;
        wcvt_kernel<<<(unsigned)((tot + 255) / 256), 256, 0, stream>>>(s, d, K, tot);
    };
    cvt(eW0, eW0t, 768); cvt(eW1, eW1t, 256); cvt(eW2, eW2t, 256);
    cvt(nW0, nW0t, 512); cvt(nW1, nW1t, 256); cvt(nW2, nW2t, 256);

    const long long* rowI = eidx;
    const long long* colI = eidx + N_EDGES;
    long long aggN = (long long)N_NODES * DIM;

    unsigned eBlocks = (N_EDGES + 63) / 64;
    unsigned nBlocks = (N_NODES + 63) / 64;

    for (int i = 0; i < 9; ++i) {
        zero_kernel<<<(unsigned)((aggN + 255) / 256), 256, 0, stream>>>(agg, aggN);
        edge_kernel<<<eBlocks, 256, SMEM_BYTES, stream>>>(
            xbuf, ea, rowI, colI,
            eW0t + (size_t)i * 768 * 256, eb0 + i * 256,
            eW1t + (size_t)i * 256 * 256, eb1 + i * 256,
            eW2t + (size_t)i * 256 * 256, eb2 + i * 256,
            eg + i * 256, ebt + i * 256, agg);
        node_kernel<<<nBlocks, 256, SMEM_BYTES, stream>>>(
            xbuf, agg,
            nW0t + (size_t)i * 512 * 256, nb0 + i * 256,
            nW1t + (size_t)i * 256 * 256, nb1 + i * 256,
            nW2t + (size_t)i * 256 * 256, nb2 + i * 256,
            ng + i * 256, nbt + i * 256);
    }
}
